// TemporalGraphGAT_47553877901451
// MI455X (gfx1250) — matrix operand order repaired and checked by reading, not compiled, not run on hardware
//
#include <hip/hip_runtime.h>
#include <hip/hip_bf16.h>
#include <math.h>

#define N_NODES 30000
#define N_EDGES 480000
#define IN_DIM  768
#define HID     128
#define HEADS   4
#define NEG_SLOPE 0.2f
#define N_ETOT  (N_EDGES + N_NODES)   // edges + self loops

// ---------------- WMMA types (CDNA5 / gfx1250, wave32) ----------------
typedef __bf16 v16bf __attribute__((ext_vector_type(16)));
typedef float  v8f   __attribute__((ext_vector_type(8)));

union BFrag { v16bf v; uint4 u[2]; };

__device__ __forceinline__ unsigned short f2bf(float f) {
  unsigned int u = __float_as_uint(f);
  unsigned int r = (u + 0x7FFFu + ((u >> 16) & 1u)) >> 16;   // RNE
  return (unsigned short)r;
}

// ---------------- conversion / transpose kernels ----------------
__global__ void f32_to_bf16_kernel(const float* __restrict__ in,
                                   unsigned short* __restrict__ out, size_t n) {
  size_t i = (size_t)blockIdx.x * blockDim.x + threadIdx.x;
  if (i < n) out[i] = f2bf(in[i]);
}

// in: K x Nc (row-major f32) -> out: Nc x K (row-major bf16)
__global__ void transpose_bf16_kernel(const float* __restrict__ in,
                                      unsigned short* __restrict__ out,
                                      int K, int Nc) {
  int i = blockIdx.x * blockDim.x + threadIdx.x;
  if (i >= K * Nc) return;
  int k = i / Nc, n = i % Nc;
  out[(size_t)n * K + k] = f2bf(in[i]);
}

__global__ void fill_kernel(float* __restrict__ p, float v, size_t n) {
  size_t i = (size_t)blockIdx.x * blockDim.x + threadIdx.x;
  if (i < n) p[i] = v;
}

// ---------------- bf16 WMMA GEMM: C[M,Nc] = A[M,K] @ B[K,Nc] ----------------
// A: M x K bf16 row-major.  BT: Nc x K bf16 (B transposed).  C: M x Nc f32.
// One wave -> 16(M) x 64(N) strip = 4 WMMA tiles, f32 accumulate.
// Software-pipelined double buffering across 32-wide K-steps: the WMMAs of
// step k consume fragments whose loads were issued one step earlier, while
// the next step's 10 loads are in flight -> partial s_wait_loadcnt, real
// load/matrix overlap inside a single wave. K must be a multiple of 64.
__global__ void wmma_gemm_bf16(const unsigned short* __restrict__ A,
                               const unsigned short* __restrict__ BT,
                               float* __restrict__ C,
                               int M, int K, int Nc) {
  int tid  = blockIdx.x * blockDim.x + threadIdx.x;
  int wid  = tid >> 5;
  int lane = tid & 31;
  int strips_m = M >> 4;
  int strips_n = Nc >> 6;
  if (wid >= strips_m * strips_n) return;
  int m0 = (wid % strips_m) << 4;
  int n0 = (wid / strips_m) << 6;

  int l15 = lane & 15;
  int hi  = (lane >> 4) & 1;

  v8f acc[4];
#pragma unroll
  for (int t = 0; t < 4; t++)
#pragma unroll
    for (int r = 0; r < 8; r++) acc[t][r] = 0.0f;

  // A fragment base: lane<16: K 0..15;
  // lane>=16: K 16..31 (contiguous, matches B)
  const unsigned short* Arow = A + (size_t)(m0 + l15) * K + hi * 16;
  // B fragment bases: lane holds 16 contiguous K for column n0+t*16+(lane&15)
  const unsigned short* Brow0 = BT + (size_t)(n0 + 0 * 16 + l15) * K + hi * 16;
  const unsigned short* Brow1 = BT + (size_t)(n0 + 1 * 16 + l15) * K + hi * 16;
  const unsigned short* Brow2 = BT + (size_t)(n0 + 2 * 16 + l15) * K + hi * 16;
  const unsigned short* Brow3 = BT + (size_t)(n0 + 3 * 16 + l15) * K + hi * 16;

  BFrag fa0, fa1, fb0[4], fb1[4];

#define LOAD_STEP(FA, FB, KK)                                         \
  do {                                                                \
    (FA).u[0] = *(const uint4*)(Arow + (KK));                         \
    (FA).u[1] = *(const uint4*)(Arow + (KK) + 8);                     \
    (FB)[0].u[0] = *(const uint4*)(Brow0 + (KK));                     \
    (FB)[0].u[1] = *(const uint4*)(Brow0 + (KK) + 8);                 \
    (FB)[1].u[0] = *(const uint4*)(Brow1 + (KK));                     \
    (FB)[1].u[1] = *(const uint4*)(Brow1 + (KK) + 8);                 \
    (FB)[2].u[0] = *(const uint4*)(Brow2 + (KK));                     \
    (FB)[2].u[1] = *(const uint4*)(Brow2 + (KK) + 8);                 \
    (FB)[3].u[0] = *(const uint4*)(Brow3 + (KK));                     \
    (FB)[3].u[1] = *(const uint4*)(Brow3 + (KK) + 8);                 \
  } while (0)

#define MMA_STEP(FA, FB)                                              \
  do {                                                                \
    acc[0] = __builtin_amdgcn_wmma_f32_16x16x32_bf16(                 \
        false, (FA).v, false, (FB)[0].v, (short)0, acc[0], false, false); \
    acc[1] = __builtin_amdgcn_wmma_f32_16x16x32_bf16(                 \
        false, (FA).v, false, (FB)[1].v, (short)0, acc[1], false, false); \
    acc[2] = __builtin_amdgcn_wmma_f32_16x16x32_bf16(                 \
        false, (FA).v, false, (FB)[2].v, (short)0, acc[2], false, false); \
    acc[3] = __builtin_amdgcn_wmma_f32_16x16x32_bf16(                 \
        false, (FA).v, false, (FB)[3].v, (short)0, acc[3], false, false); \
  } while (0)

  // prologue: K-step 0 in flight
  LOAD_STEP(fa0, fb0, 0);

#pragma unroll 1
  for (int k0 = 0; k0 < K - 64; k0 += 64) {
    LOAD_STEP(fa1, fb1, k0 + 32);   // issue next step's loads first
    MMA_STEP(fa0, fb0);             // consume step issued one iteration ago
    LOAD_STEP(fa0, fb0, k0 + 64);
    MMA_STEP(fa1, fb1);
  }
  // epilogue: last 64 K (fb0 holds K-64, issued in final loop iteration)
  LOAD_STEP(fa1, fb1, K - 32);
  MMA_STEP(fa0, fb0);
  MMA_STEP(fa1, fb1);

#undef LOAD_STEP
#undef MMA_STEP

  // C/D layout: VGPR r, lanes 0-15 -> M=r, lanes 16-31 -> M=8+r; N = n0+t*16+(lane&15)
#pragma unroll
  for (int t = 0; t < 4; t++) {
    int col = n0 + t * 16 + l15;
#pragma unroll
    for (int r = 0; r < 8; r++) {
      int row = m0 + r + (hi << 3);
      C[(size_t)row * Nc + col] = acc[t][r];
    }
  }
}

// ---------------- attention coefficients ----------------
// as_out[n,h] = dot(h[n,h,:], a_s[h,:]) ; ad_out likewise
__global__ void alpha_kernel(const float* __restrict__ h,
                             const float* __restrict__ a_s,
                             const float* __restrict__ a_d,
                             float* __restrict__ as_out, float* __restrict__ ad_out,
                             int n, int Hh, int C) {
  int i = blockIdx.x * blockDim.x + threadIdx.x;
  if (i >= n * Hh) return;
  int node = i / Hh, hd = i % Hh;
  const float* hp = h + (size_t)node * Hh * C + (size_t)hd * C;
  const float* sp = a_s + (size_t)hd * C;
  const float* dp = a_d + (size_t)hd * C;
  float ss = 0.f, sd = 0.f;
  for (int c = 0; c < C; c++) { float v = hp[c]; ss += v * sp[c]; sd += v * dp[c]; }
  as_out[i] = ss;
  ad_out[i] = sd;
}

// ---------------- edge kernels (segment softmax + scatter-add) ----------------
__device__ __forceinline__ void atomicMaxF(float* addr, float val) {
  if (val >= 0.f) atomicMax((int*)addr, __float_as_int(val));
  else            atomicMin((unsigned int*)addr, __float_as_uint(val));
}

__device__ __forceinline__ void edge_sd(const int* __restrict__ ei, int eidx, int& s, int& d) {
  if (eidx < N_EDGES) { s = ei[eidx]; d = ei[N_EDGES + eidx]; }
  else                { s = d = eidx - N_EDGES; }              // self loop
}

__global__ void edge_max_kernel(const int* __restrict__ ei,
                                const float* __restrict__ as_, const float* __restrict__ ad_,
                                float* __restrict__ e_out, float* __restrict__ m_out, int Hh) {
  int i = blockIdx.x * blockDim.x + threadIdx.x;
  if (i >= N_ETOT * Hh) return;
  int eidx = i / Hh, hd = i % Hh, s, d;
  edge_sd(ei, eidx, s, d);
  float e = as_[s * Hh + hd] + ad_[d * Hh + hd];
  e = e > 0.f ? e : NEG_SLOPE * e;                              // leaky_relu
  e_out[i] = e;
  atomicMaxF(&m_out[d * Hh + hd], e);
}

__global__ void edge_ex_kernel(const int* __restrict__ ei,
                               const float* __restrict__ m,
                               float* __restrict__ e_inout, float* __restrict__ denom, int Hh) {
  int i = blockIdx.x * blockDim.x + threadIdx.x;
  if (i >= N_ETOT * Hh) return;
  int eidx = i / Hh, hd = i % Hh, s, d;
  edge_sd(ei, eidx, s, d);
  float ex = __expf(e_inout[i] - m[d * Hh + hd]);
  e_inout[i] = ex;
  atomicAdd(&denom[d * Hh + hd], ex);
}

__global__ void edge_agg_kernel(const int* __restrict__ ei,
                                const float* __restrict__ ex,
                                const float* __restrict__ h,
                                float* __restrict__ agg, int Hh, int C) {
  long long i = (long long)blockIdx.x * blockDim.x + threadIdx.x;
  int HC = Hh * C;
  if (i >= (long long)N_ETOT * HC) return;
  int eidx = (int)(i / HC);
  int rem  = (int)(i % HC);
  int hd = rem / C, s, d;
  edge_sd(ei, eidx, s, d);
  atomicAdd(&agg[(size_t)d * HC + rem],
            ex[(size_t)eidx * Hh + hd] * h[(size_t)s * HC + rem]);
}

// out = elu(agg/denom + b); optional f32 and bf16 outputs
__global__ void finalize_kernel(const float* __restrict__ agg,
                                const float* __restrict__ denom,
                                const float* __restrict__ b,
                                float* __restrict__ outf,
                                unsigned short* __restrict__ outb,
                                int n, int Hh, int C) {
  int i = blockIdx.x * blockDim.x + threadIdx.x;
  int HC = Hh * C;
  if (i >= n * HC) return;
  int node = i / HC, rem = i % HC, hd = rem / C;
  float v = agg[i] / (denom[node * Hh + hd] + 1e-16f) + b[rem];
  v = v > 0.f ? v : (__expf(v) - 1.f);                          // elu
  if (outf) outf[i] = v;
  if (outb) outb[i] = f2bf(v);
}

// ---------------- final reduction + FC ----------------
__global__ void colmean_kernel(const float* __restrict__ h2, float* __restrict__ mean,
                               int n, int C) {
  __shared__ float sdata[256];
  int c = blockIdx.x;
  float s = 0.f;
  for (int r = threadIdx.x; r < n; r += blockDim.x) s += h2[(size_t)r * C + c];
  sdata[threadIdx.x] = s;
  __syncthreads();
  for (int off = 128; off > 0; off >>= 1) {
    if ((int)threadIdx.x < off) sdata[threadIdx.x] += sdata[threadIdx.x + off];
    __syncthreads();
  }
  if (threadIdx.x == 0) mean[c] = sdata[0] / (float)n;
}

__global__ void final_fc_kernel(const float* __restrict__ mean,
                                const float* __restrict__ fcW,
                                const float* __restrict__ fcb,
                                float* __restrict__ out) {
  int j = blockIdx.x * blockDim.x + threadIdx.x;
  if (j >= IN_DIM) return;
  float s = fcb[j];
  for (int c = 0; c < HID; c++) s += mean[c] * fcW[c * IN_DIM + j];
  out[j] = s;
}

// ---------------- launcher ----------------
extern "C" void kernel_launch(void* const* d_in, const int* in_sizes, int n_in,
                              void* d_out, int out_size, void* d_ws, size_t ws_size,
                              hipStream_t stream) {
  (void)in_sizes; (void)n_in; (void)out_size; (void)ws_size;
  const float* x      = (const float*)d_in[0];
  const int*   ei     = (const int*)  d_in[1];
  const float* W1     = (const float*)d_in[2];
  const float* a_src1 = (const float*)d_in[3];
  const float* a_dst1 = (const float*)d_in[4];
  const float* b1     = (const float*)d_in[5];
  const float* W2     = (const float*)d_in[6];
  const float* a_src2 = (const float*)d_in[7];
  const float* a_dst2 = (const float*)d_in[8];
  const float* b2     = (const float*)d_in[9];
  const float* fcW    = (const float*)d_in[10];
  const float* fcb    = (const float*)d_in[11];
  float* out = (float*)d_out;

  const int HC1 = HEADS * HID;   // 512
  auto al = [](size_t x) { return (x + 255) & ~(size_t)255; };

  // workspace layout (with reuse; peak ~211 MB)
  size_t o_xb   = 0;                                    // N*IN bf16 (GEMM1 A)
  size_t o_w1t  = al(o_xb  + (size_t)N_NODES * IN_DIM * 2);
  size_t o_h1   = al(o_w1t + (size_t)IN_DIM * HC1 * 2); // N*512 f32
  size_t o_as1  = al(o_h1  + (size_t)N_NODES * HC1 * 4);
  size_t o_ad1  = al(o_as1 + (size_t)N_NODES * HEADS * 4);
  size_t o_e    = al(o_ad1 + (size_t)N_NODES * HEADS * 4);
  size_t o_m    = al(o_e   + (size_t)N_ETOT * HEADS * 4);
  size_t o_den  = al(o_m   + (size_t)N_NODES * HEADS * 4);
  size_t o_agg1 = al(o_den + (size_t)N_NODES * HEADS * 4);
  size_t o_h1b  = al(o_agg1 + (size_t)N_NODES * HC1 * 4); // N*512 bf16 (GEMM2 A)
  size_t o_mean = al(o_h1b  + (size_t)N_NODES * HC1 * 2);
  // dead-buffer reuse for layer 2
  size_t o_h2pre = o_xb;                   // N*128 f32 (fits in xb region)
  size_t o_w2t   = o_w1t;                  // 128x512 bf16
  size_t o_agg2  = o_h1;                   // N*128 f32
  size_t o_h2act = o_h1 + (16u << 20);     // N*128 f32, still inside h1 region

  char* W = (char*)d_ws;
  unsigned short* xb    = (unsigned short*)(W + o_xb);
  unsigned short* W1T   = (unsigned short*)(W + o_w1t);
  float*          h1    = (float*)(W + o_h1);
  float*          as1   = (float*)(W + o_as1);
  float*          ad1   = (float*)(W + o_ad1);
  float*          ebuf  = (float*)(W + o_e);
  float*          mbuf  = (float*)(W + o_m);
  float*          den   = (float*)(W + o_den);
  float*          agg1  = (float*)(W + o_agg1);
  unsigned short* h1b   = (unsigned short*)(W + o_h1b);
  float*          meanv = (float*)(W + o_mean);
  float*          h2pre = (float*)(W + o_h2pre);
  unsigned short* W2T   = (unsigned short*)(W + o_w2t);
  float*          agg2  = (float*)(W + o_agg2);
  float*          h2act = (float*)(W + o_h2act);

  auto cdiv = [](long long a, long long b) { return (int)((a + b - 1) / b); };
  const int BT = 256;

  // ---- layer 1 ----
  size_t nx = (size_t)N_NODES * IN_DIM;
  f32_to_bf16_kernel<<<cdiv((long long)nx, BT), BT, 0, stream>>>(x, xb, nx);
  transpose_bf16_kernel<<<cdiv((long long)IN_DIM * HC1, BT), BT, 0, stream>>>(W1, W1T, IN_DIM, HC1);
  {
    int waves = (N_NODES / 16) * (HC1 / 64);
    wmma_gemm_bf16<<<cdiv((long long)waves * 32, BT), BT, 0, stream>>>(xb, W1T, h1, N_NODES, IN_DIM, HC1);
  }
  alpha_kernel<<<cdiv((long long)N_NODES * HEADS, BT), BT, 0, stream>>>(
      h1, a_src1, a_dst1, as1, ad1, N_NODES, HEADS, HID);
  fill_kernel<<<cdiv((long long)N_NODES * HEADS, BT), BT, 0, stream>>>(mbuf, -3.0e38f, (size_t)N_NODES * HEADS);
  fill_kernel<<<cdiv((long long)N_NODES * HEADS, BT), BT, 0, stream>>>(den, 0.f, (size_t)N_NODES * HEADS);
  fill_kernel<<<cdiv((long long)N_NODES * HC1, BT), BT, 0, stream>>>(agg1, 0.f, (size_t)N_NODES * HC1);
  edge_max_kernel<<<cdiv((long long)N_ETOT * HEADS, BT), BT, 0, stream>>>(ei, as1, ad1, ebuf, mbuf, HEADS);
  edge_ex_kernel<<<cdiv((long long)N_ETOT * HEADS, BT), BT, 0, stream>>>(ei, mbuf, ebuf, den, HEADS);
  edge_agg_kernel<<<cdiv((long long)N_ETOT * HC1, BT), BT, 0, stream>>>(ei, ebuf, h1, agg1, HEADS, HID);
  finalize_kernel<<<cdiv((long long)N_NODES * HC1, BT), BT, 0, stream>>>(
      agg1, den, b1, (float*)nullptr, h1b, N_NODES, HEADS, HID);

  // ---- layer 2 ----
  transpose_bf16_kernel<<<cdiv((long long)HC1 * HID, BT), BT, 0, stream>>>(W2, W2T, HC1, HID);
  {
    int waves = (N_NODES / 16) * (HID / 64);
    wmma_gemm_bf16<<<cdiv((long long)waves * 32, BT), BT, 0, stream>>>(h1b, W2T, h2pre, N_NODES, HC1, HID);
  }
  alpha_kernel<<<cdiv((long long)N_NODES, BT), BT, 0, stream>>>(
      h2pre, a_src2, a_dst2, as1, ad1, N_NODES, 1, HID);
  fill_kernel<<<cdiv((long long)N_NODES, BT), BT, 0, stream>>>(mbuf, -3.0e38f, (size_t)N_NODES);
  fill_kernel<<<cdiv((long long)N_NODES, BT), BT, 0, stream>>>(den, 0.f, (size_t)N_NODES);
  fill_kernel<<<cdiv((long long)N_NODES * HID, BT), BT, 0, stream>>>(agg2, 0.f, (size_t)N_NODES * HID);
  edge_max_kernel<<<cdiv((long long)N_ETOT, BT), BT, 0, stream>>>(ei, as1, ad1, ebuf, mbuf, 1);
  edge_ex_kernel<<<cdiv((long long)N_ETOT, BT), BT, 0, stream>>>(ei, mbuf, ebuf, den, 1);
  edge_agg_kernel<<<cdiv((long long)N_ETOT * HID, BT), BT, 0, stream>>>(ei, ebuf, h2pre, agg2, 1, HID);
  finalize_kernel<<<cdiv((long long)N_NODES * HID, BT), BT, 0, stream>>>(
      agg2, den, b2, h2act, (unsigned short*)nullptr, N_NODES, 1, HID);

  // ---- mean over nodes + final FC ----
  colmean_kernel<<<HID, 256, 0, stream>>>(h2act, meanv, N_NODES, HID);
  final_fc_kernel<<<cdiv(IN_DIM, BT), BT, 0, stream>>>(meanv, fcW, fcb, out);
}
